// MyConvLSTMCell_39831526703814
// MI455X (gfx1250) — compile-verified, matrix operand order checked by reading, not run on hardware
//
#include <hip/hip_runtime.h>

#define SQ   8192
#define DM   512
#define DK   16
#define DFF  2048

typedef __attribute__((ext_vector_type(16))) __bf16 v16bf;
typedef __attribute__((ext_vector_type(8)))  __bf16 v8bf;
typedef __attribute__((ext_vector_type(8)))  float  v8f;

// ---------------------------------------------------------------------------
// D = A(16x32 bf16) * B(32x16 bf16) + C(16x16 f32)
// ---------------------------------------------------------------------------
__device__ __forceinline__ v8f wmma_bf16(v16bf a, v16bf b, v8f c) {
  return __builtin_amdgcn_wmma_f32_16x16x32_bf16(
      /*neg_a=*/false, a, /*neg_b=*/false, b,
      /*c_mod=*/(short)0, c, /*reuse_a=*/false, /*reuse_b=*/false);
}

// Load a 16x32 bf16 A-fragment (or 32x16 B-fragment from N-major storage).
// ISA layout (CDNA5 ISA §7.12.2):
//   lanes 0-15 : row = lane,    elems 0-7 = K 0-7,  elems 8-15 = K 16-23
//   lanes 16-31: row = lane-16, elems 0-7 = K 8-15, elems 8-15 = K 24-31
__device__ __forceinline__ v16bf load_frag(const __bf16* base, int ld) {
  const int lane = threadIdx.x & 31;
  const int row  = lane & 15;
  const int kb   = (lane >> 4) << 3;      // 0 or 8
  const __bf16* p = base + row * ld + kb;
  v8bf lo = *(const v8bf*)(p);
  v8bf hi = *(const v8bf*)(p + 16);
  v16bf o;
#pragma unroll
  for (int i = 0; i < 8; ++i) { o[i] = lo[i]; o[i + 8] = hi[i]; }
  return o;
}

// ---------------------------------------------------------------------------
// CDNA5 async copy: global -> LDS through the async engine (ASYNCcnt).
// LDS destination VGPR holds the 32-bit LDS byte offset; per the ISA the
// generic pointer's low 32 bits are exactly that offset.
// ---------------------------------------------------------------------------
__device__ __forceinline__ void async_ld16(unsigned lds_off, const __bf16* gsrc) {
  asm volatile("global_load_async_to_lds_b128 %0, %1, off"
               :: "v"(lds_off), "v"(gsrc)
               : "memory");
}
template <int N>
__device__ __forceinline__ void wait_async() {
  asm volatile("s_wait_asynccnt %0" :: "i"(N) : "memory");
}

// ---------------------------------------------------------------------------
// Prep: fp32 -> bf16 conversions, weight transposes, summed W_o.
// ---------------------------------------------------------------------------
__global__ __launch_bounds__(256)
void prep_kernel(const float* __restrict__ frame, const float* __restrict__ Wq,
                 const float* __restrict__ Wk, const float* __restrict__ Wv,
                 const float* __restrict__ Wo, const float* __restrict__ fc1w,
                 const float* __restrict__ fc2w,
                 __bf16* frame_bf, __bf16* Wq_t, __bf16* Wk_t, __bf16* Wv_t,
                 __bf16* Wsum_t, __bf16* fc1_wb, __bf16* fc2_wb) {
  const int tid  = blockIdx.x * blockDim.x + threadIdx.x;
  const int nthr = gridDim.x * blockDim.x;
  for (int i = tid; i < SQ * DM; i += nthr) frame_bf[i] = (__bf16)frame[i];
  for (int i = tid; i < DK * DM; i += nthr) {      // W[k][n] -> Wt[n][k]
    int n = i / DM, k = i % DM;
    Wq_t[n * DM + k] = (__bf16)Wq[k * DK + n];
    Wk_t[n * DM + k] = (__bf16)Wk[k * DK + n];
    Wv_t[n * DM + k] = (__bf16)Wv[k * DK + n];
  }
  for (int i = tid; i < DM * 32; i += nthr) {      // Wsum_t[n][dv], K padded to 32
    int n = i >> 5, k = i & 31;
    float s = 0.f;
    if (k < DK)
      for (int h = 0; h < 8; ++h) s += Wo[(h * DK + k) * DM + n];
    Wsum_t[n * 32 + k] = (__bf16)s;
  }
  for (int i = tid; i < DFF * DM; i += nthr) fc1_wb[i] = (__bf16)fc1w[i];
  for (int i = tid; i < DM * DFF; i += nthr) fc2_wb[i] = (__bf16)fc2w[i];
}

// ---------------------------------------------------------------------------
// QKV projection: one wave per 16-row tile. Q scaled by 1/sqrt(dk)=0.25.
// Q,K stored [S,32] row-major (dk padded to 32 with zeros); V stored
// transposed Vt[16][S] so attention's second WMMA reads it as an A-fragment.
// ---------------------------------------------------------------------------
__global__ __launch_bounds__(128)
void qkv_kernel(const __bf16* __restrict__ frame_bf, const __bf16* __restrict__ Wq_t,
                const __bf16* __restrict__ Wk_t, const __bf16* __restrict__ Wv_t,
                __bf16* __restrict__ Qb, __bf16* __restrict__ Kb, __bf16* __restrict__ Vt) {
  const int wave = (blockIdx.x * blockDim.x + threadIdx.x) >> 5;   // 0..511
  const int m0 = wave * 16;
  v8f aq = {}, ak = {}, av = {};
  for (int kk = 0; kk < DM; kk += 32) {
    v16bf a  = load_frag(frame_bf + m0 * DM + kk, DM);
    v16bf bq = load_frag(Wq_t + kk, DM);
    v16bf bk = load_frag(Wk_t + kk, DM);
    v16bf bv = load_frag(Wv_t + kk, DM);
    aq = wmma_bf16(a, bq, aq);
    ak = wmma_bf16(a, bk, ak);
    av = wmma_bf16(a, bv, av);
  }
  const int lane = threadIdx.x & 31;
  const int n  = lane & 15;              // output col (dk index)
  const int mb = (lane >> 4) << 3;       // row base 0 or 8
#pragma unroll
  for (int r = 0; r < 8; ++r) {
    int gm = m0 + mb + r;
    Qb[gm * 32 + n]      = (__bf16)(aq[r] * 0.25f);
    Qb[gm * 32 + 16 + n] = (__bf16)0.f;
    Kb[gm * 32 + n]      = (__bf16)ak[r];
    Kb[gm * 32 + 16 + n] = (__bf16)0.f;
    Vt[n * SQ + gm]      = (__bf16)av[r];
  }
}

// ---------------------------------------------------------------------------
// Flash attention, one wave per 16 query rows.
// S^T tiles (K @ Q^T) put the query index in the lane dimension, so online
// softmax state (m,l) is per-lane; P^T lands directly in the B-fragment
// layout for head^T += V^T @ P^T. Only cross-lane op: shfl_xor(16).
// ---------------------------------------------------------------------------
__global__ __launch_bounds__(128)
void attn_kernel(const __bf16* __restrict__ Qb, const __bf16* __restrict__ Kb,
                 const __bf16* __restrict__ Vt, __bf16* __restrict__ head_b) {
  const int wave = (blockIdx.x * blockDim.x + threadIdx.x) >> 5;   // 0..511
  const int q0 = wave * 16;
  const int lane = threadIdx.x & 31;
  const v16bf bq = load_frag(Qb + q0 * 32, 32);    // B: [d=32] x [n=q=16]
  float m_run = -1e30f, l_run = 0.f;
  v8f acc = {};                                    // head^T tile: M=dv, N=q
  for (int j0 = 0; j0 < SQ; j0 += 32) {
    v16bf ak0 = load_frag(Kb + j0 * 32, 32);       // keys j0..j0+15
    v16bf ak1 = load_frag(Kb + (j0 + 16) * 32, 32);
    v8f z = {};
    v8f t0 = wmma_bf16(ak0, bq, z);                // S^T[kc=0..15][q]
    v8f t1 = wmma_bf16(ak1, bq, z);                // S^T[kc=16..31][q]
    float mt = -1e30f;
#pragma unroll
    for (int i = 0; i < 8; ++i) mt = fmaxf(mt, fmaxf(t0[i], t1[i]));
    mt = fmaxf(mt, __shfl_xor(mt, 16, 32));        // full 32-key row max for q
    float mnew  = fmaxf(m_run, mt);
    float alpha = __expf(m_run - mnew);
    float lsum = 0.f;
    v16bf bp;                                      // P^T as B-fragment
#pragma unroll
    for (int i = 0; i < 8; ++i) {
      float p0 = __expf(t0[i] - mnew);
      float p1 = __expf(t1[i] - mnew);
      lsum += p0 + p1;
      bp[i]     = (__bf16)p0;
      bp[i + 8] = (__bf16)p1;
    }
    lsum += __shfl_xor(lsum, 16, 32);
    l_run = l_run * alpha + lsum;
    m_run = mnew;
#pragma unroll
    for (int r = 0; r < 8; ++r) acc[r] *= alpha;   // per-lane rescale
    v16bf av = load_frag(Vt + j0, SQ);             // A: V^T[dv][kc]
    acc = wmma_bf16(av, bp, acc);                  // head^T += V^T @ P^T
  }
  const float inv = 1.f / l_run;
  const int q   = lane & 15;
  const int dvb = (lane >> 4) << 3;
#pragma unroll
  for (int r = 0; r < 8; ++r) {
    int dv = dvb + r;
    head_b[(q0 + q) * 32 + dv]      = (__bf16)(acc[r] * inv);
    head_b[(q0 + q) * 32 + 16 + dv] = (__bf16)0.f;   // zero pad (K=32)
  }
}

// ---------------------------------------------------------------------------
// mho = head @ Wsum  ([8192,16(pad32)] @ [16,512]); one WMMA per 16x16 tile.
// Writes both f32 (residual) and bf16 (fc1 input).
// ---------------------------------------------------------------------------
__global__ __launch_bounds__(256)
void mho_kernel(const __bf16* __restrict__ head_b, const __bf16* __restrict__ Wsum_t,
                float* __restrict__ mho_f, __bf16* __restrict__ mho_bf) {
  const int wave = (blockIdx.x * blockDim.x + threadIdx.x) >> 5;   // 0..16383
  const int mt = wave >> 5;   // 0..511
  const int nt = wave & 31;   // 0..31
  v16bf a = load_frag(head_b + mt * 16 * 32, 32);
  v16bf b = load_frag(Wsum_t + nt * 16 * 32, 32);
  v8f c = {};
  c = wmma_bf16(a, b, c);
  const int lane = threadIdx.x & 31;
  const int n  = lane & 15;
  const int mb = (lane >> 4) << 3;
#pragma unroll
  for (int r = 0; r < 8; ++r) {
    int gm = mt * 16 + mb + r;
    int gn = nt * 16 + n;
    float v = c[r];
    mho_f[gm * DM + gn]  = v;
    mho_bf[gm * DM + gn] = (__bf16)v;
  }
}

// ---------------------------------------------------------------------------
// Tiled GEMM: C[M,N] = A[M,K] @ B[N,K]^T (+bias, +optional GELU / residual).
// 64x64 block tile; K-panels double-buffered in LDS via async-to-LDS engine
// (global_load_async_to_lds_b128 + s_wait_asynccnt). 4 waves x (2x2) WMMAs.
// ---------------------------------------------------------------------------
template <int DO_GELU, int K>
__global__ __launch_bounds__(128)
void gemm_kernel(const __bf16* __restrict__ A, const __bf16* __restrict__ B,
                 const float* __restrict__ bias, const float* __restrict__ residual,
                 float* __restrict__ outF, __bf16* __restrict__ outB, int N) {
  __shared__ __bf16 As[2][64][40];
  __shared__ __bf16 Bs[2][64][40];
  const int m0 = blockIdx.x * 64;
  const int n0 = blockIdx.y * 64;
  const int t  = threadIdx.x;
  const int wave = t >> 5;
  const int wm = (wave >> 1) * 32;
  const int wn = (wave & 1) * 32;

  // This thread's two 16B chunks of each 64x32 panel.
  const int c0   = t * 2;
  const int row0 = c0 >> 2,        col0 = (c0 & 3) << 3;
  const int row1 = (c0 + 1) >> 2,  col1 = ((c0 + 1) & 3) << 3;

  auto issue = [&](int buf, int kk) {
    async_ld16((unsigned)(uintptr_t)&As[buf][row0][col0],
               A + (size_t)(m0 + row0) * K + kk + col0);
    async_ld16((unsigned)(uintptr_t)&As[buf][row1][col1],
               A + (size_t)(m0 + row1) * K + kk + col1);
    async_ld16((unsigned)(uintptr_t)&Bs[buf][row0][col0],
               B + (size_t)(n0 + row0) * K + kk + col0);
    async_ld16((unsigned)(uintptr_t)&Bs[buf][row1][col1],
               B + (size_t)(n0 + row1) * K + kk + col1);
  };

  v8f acc[2][2] = {};
  issue(0, 0);
  for (int kk = 0; kk < K; kk += 32) {
    const int cur = (kk >> 5) & 1;
    if (kk + 32 < K) {
      issue(cur ^ 1, kk + 32);   // prefetch next panel into other buffer
      wait_async<4>();           // 4 oldest (current panel) complete, in order
    } else {
      wait_async<0>();
    }
    __syncthreads();             // current panel visible to all waves
    v16bf af[2], bfrag[2];
#pragma unroll
    for (int i = 0; i < 2; ++i) {
      af[i]    = load_frag(&As[cur][wm + i * 16][0], 40);
      bfrag[i] = load_frag(&Bs[cur][wn + i * 16][0], 40);
    }
#pragma unroll
    for (int mi = 0; mi < 2; ++mi)
#pragma unroll
      for (int ni = 0; ni < 2; ++ni)
        acc[mi][ni] = wmma_bf16(af[mi], bfrag[ni], acc[mi][ni]);
    __syncthreads();             // done reading before buffer is refilled
  }
  const int lane = t & 31;
  const int cn  = lane & 15;
  const int cmb = (lane >> 4) << 3;
#pragma unroll
  for (int mi = 0; mi < 2; ++mi)
#pragma unroll
    for (int ni = 0; ni < 2; ++ni)
#pragma unroll
      for (int r = 0; r < 8; ++r) {
        int gm = m0 + wm + mi * 16 + cmb + r;
        int gn = n0 + wn + ni * 16 + cn;
        float v = acc[mi][ni][r] + bias[gn];
        if (DO_GELU) {
          v = 0.5f * v * (1.f + erff(v * 0.70710678118f));   // exact-erf GELU
          outB[(size_t)gm * N + gn] = (__bf16)v;
        } else {
          outF[(size_t)gm * N + gn] = v + residual[(size_t)gm * N + gn];
        }
      }
}

// ---------------------------------------------------------------------------
extern "C" void kernel_launch(void* const* d_in, const int* in_sizes, int n_in,
                              void* d_out, int out_size, void* d_ws, size_t ws_size,
                              hipStream_t stream) {
  (void)in_sizes; (void)n_in; (void)out_size; (void)ws_size;
  const float* frame = (const float*)d_in[0];
  const float* Wq    = (const float*)d_in[1];
  const float* Wk    = (const float*)d_in[2];
  const float* Wv    = (const float*)d_in[3];
  const float* Wo    = (const float*)d_in[4];
  const float* fc1w  = (const float*)d_in[5];
  const float* fc1b  = (const float*)d_in[6];
  const float* fc2w  = (const float*)d_in[7];
  const float* fc2b  = (const float*)d_in[8];
  float* out = (float*)d_out;

  char* ws = (char*)d_ws;
  size_t off = 0;
  auto alloc = [&](size_t bytes) -> char* {
    char* p = ws + off;
    off += (bytes + 255) & ~(size_t)255;
    return p;
  };
  __bf16* frame_bf = (__bf16*)alloc((size_t)SQ * DM * 2);
  __bf16* Qb       = (__bf16*)alloc((size_t)SQ * 32 * 2);
  __bf16* Kb       = (__bf16*)alloc((size_t)SQ * 32 * 2);
  __bf16* Vt       = (__bf16*)alloc((size_t)DK * SQ * 2);
  __bf16* head_b   = (__bf16*)alloc((size_t)SQ * 32 * 2);
  float*  mho_f    = (float*) alloc((size_t)SQ * DM * 4);
  __bf16* mho_bf   = (__bf16*)alloc((size_t)SQ * DM * 2);
  __bf16* h_bf     = (__bf16*)alloc((size_t)SQ * DFF * 2);
  __bf16* Wq_t     = (__bf16*)alloc((size_t)DK * DM * 2);
  __bf16* Wk_t     = (__bf16*)alloc((size_t)DK * DM * 2);
  __bf16* Wv_t     = (__bf16*)alloc((size_t)DK * DM * 2);
  __bf16* Wsum_t   = (__bf16*)alloc((size_t)DM * 32 * 2);
  __bf16* fc1_wb   = (__bf16*)alloc((size_t)DFF * DM * 2);
  __bf16* fc2_wb   = (__bf16*)alloc((size_t)DM * DFF * 2);

  prep_kernel<<<2048, 256, 0, stream>>>(frame, Wq, Wk, Wv, Wo, fc1w, fc2w,
                                        frame_bf, Wq_t, Wk_t, Wv_t, Wsum_t,
                                        fc1_wb, fc2_wb);
  qkv_kernel<<<128, 128, 0, stream>>>(frame_bf, Wq_t, Wk_t, Wv_t, Qb, Kb, Vt);
  attn_kernel<<<128, 128, 0, stream>>>(Qb, Kb, Vt, head_b);
  mho_kernel<<<2048, 256, 0, stream>>>(head_b, Wsum_t, mho_f, mho_bf);
  gemm_kernel<1, DM><<<dim3(SQ / 64, DFF / 64), 128, 0, stream>>>(
      mho_bf, fc1_wb, fc1b, nullptr, nullptr, h_bf, DFF);
  gemm_kernel<0, DFF><<<dim3(SQ / 64, DM / 64), 128, 0, stream>>>(
      h_bf, fc2_wb, fc2b, mho_f, out, nullptr, DM);
}
